// CodeBook_20143396618800
// MI455X (gfx1250) — matrix operand order repaired and checked by reading, not compiled, not run on hardware
//
#include <hip/hip_runtime.h>
#include <hip/hip_bf16.h>

// Problem constants (from reference)
#define BB   32
#define CC   256
#define HH   32
#define WW   32
#define NTOK (BB * HH * WW)   // 32768 tokens
#define KCODE 1024
#define BETA 0.25f

typedef __attribute__((ext_vector_type(16))) __bf16 v16bf;
typedef __attribute__((ext_vector_type(8)))  float  v8f;

__device__ __forceinline__ unsigned short f32_to_bf16_rne(float x) {
    unsigned int u = __float_as_uint(x);
    unsigned int r = u + 0x7FFFu + ((u >> 16) & 1u);
    return (unsigned short)(r >> 16);
}
__device__ __forceinline__ float bf16_to_f32(unsigned short h) {
    return __uint_as_float(((unsigned int)h) << 16);
}

// ---------------------------------------------------------------------------
// Prep: repack codebook [K=1024, C=256] fp32 into WMMA B-fragment layout,
// split into bf16 hi/lo. B matrix is 32(k) x 16(n) bf16 per fragment:
//   lanes 0-15  -> k = 0..15 (half j = k), column n = lane
//   lanes 16-31 -> k = 16..31,             column n = lane-16
// bpack[(kt*8 + cs)*32 + lane][j], kt in [0,64) code tiles, cs in [0,8) C-steps.
// ---------------------------------------------------------------------------
__global__ void vq_prep_bpack(const float* __restrict__ codebook,
                              unsigned short* __restrict__ bpH,
                              unsigned short* __restrict__ bpL) {
    int flat = blockIdx.x * blockDim.x + threadIdx.x;   // 64*8*32*16 = 262144
    if (flat >= 64 * 8 * 32 * 16) return;
    int j    = flat & 15;
    int lane = (flat >> 4) & 31;
    int cs   = (flat >> 9) & 7;
    int kt   = flat >> 12;
    int n    = lane & 15;
    int kin  = ((lane >> 4) << 4) + j;      // k within 32-wide C step
    int c    = cs * 32 + kin;
    int code = kt * 16 + n;
    float x  = codebook[code * CC + c];
    unsigned short hi = f32_to_bf16_rne(x);
    unsigned short lo = f32_to_bf16_rne(x - bf16_to_f32(hi));
    bpH[flat] = hi;
    bpL[flat] = lo;
}

__global__ void vq_prep_wsq(const float* __restrict__ codebook,
                            float* __restrict__ wsq) {
    int code = blockIdx.x * blockDim.x + threadIdx.x;
    if (code >= KCODE) return;
    float s = 0.f;
    for (int c = 0; c < CC; ++c) {
        float v = codebook[code * CC + c];
        s += v * v;
    }
    wsq[code] = s;
}

__global__ void vq_init_loss(float* __restrict__ loss) {
    if (threadIdx.x == 0 && blockIdx.x == 0) *loss = 0.f;
}

// ---------------------------------------------------------------------------
// Main: fused GEMM + argmin.
// Block = 256 threads (8 waves). Each wave owns one 16-token row tile
// (128 tokens per block). For each of 64 code tiles (16 codes), accumulate
// dot(z, w) over C=256 via 8 C-steps x 3 bf16-split WMMAs into THREE
// independent accumulator chains (ILP for the XDL pipe), then update the
// running per-token argmin of |w|^2 - 2 z.w  (|z|^2 is constant per token).
// B tiles are double-buffered in LDS: global loads for tile kt+1 issue
// before the WMMAs of tile kt; one barrier per iteration.
// ---------------------------------------------------------------------------
__global__ void __launch_bounds__(256)
vq_quantize(const float* __restrict__ z,
            const unsigned short* __restrict__ bpH,
            const unsigned short* __restrict__ bpL,
            const float* __restrict__ wsq,
            int* __restrict__ idxOut) {
    __shared__ unsigned short bH[2][8 * 32 * 16];  // 2 x 8 KB: B hi tile
    __shared__ unsigned short bL[2][8 * 32 * 16];  // 2 x 8 KB: B lo tile
    __shared__ float          swsq[KCODE];         // 4 KB: |w_k|^2

    const int tid  = threadIdx.x;
    const int wave = tid >> 5;
    const int lane = tid & 31;
    const int t0   = blockIdx.x * 128;           // first token of block
    const int b    = t0 >> 10;                   // batch index (H*W = 1024)
    const int s0   = t0 & 1023;                  // spatial offset within batch

    // --- Stage |w|^2 into LDS (each thread copies 4 floats) ---
#pragma unroll
    for (int i = 0; i < 4; ++i) swsq[tid + 256 * i] = wsq[tid + 256 * i];

    // --- Load this wave's A fragments (16 tokens x 256 C) hi/lo into regs ---
    // 16-bit A 16x32 layout: lane&15 = token row M; lane>>4 = g selects K set;
// half j -> k = 16g + j.
    union Frag { v16bf v; unsigned short h[16]; };
    Frag ahi[8], alo[8];
    {
        const int mr = lane & 15;
        const int g  = lane >> 4;
        const float* zb = z + ((size_t)b * CC << 10) + s0 + wave * 16 + mr;
#pragma unroll
        for (int cs = 0; cs < 8; ++cs) {
#pragma unroll
            for (int j = 0; j < 16; ++j) {
                int vv = j >> 1, p = j & 1;
                int k  = 16 * g + j;
                int c  = cs * 32 + k;
                float x = zb[(size_t)c << 10];
                unsigned short hi = f32_to_bf16_rne(x);
                ahi[cs].h[j] = hi;
                alo[cs].h[j] = f32_to_bf16_rne(x - bf16_to_f32(hi));
            }
        }
    }

    // --- Prologue: stage B tile 0 into LDS buffer 0 ---
    {
        const uint4* gH = reinterpret_cast<const uint4*>(bpH);
        const uint4* gL = reinterpret_cast<const uint4*>(bpL);
        uint4* sH = reinterpret_cast<uint4*>(bH[0]);
        uint4* sL = reinterpret_cast<uint4*>(bL[0]);
        sH[tid]       = gH[tid];
        sH[tid + 256] = gH[tid + 256];
        sL[tid]       = gL[tid];
        sL[tid + 256] = gL[tid + 256];
    }
    __syncthreads();

    float bestS[8];
    int   bestI[8];
#pragma unroll
    for (int v = 0; v < 8; ++v) { bestS[v] = 3.4e38f; bestI[v] = 0; }

    for (int kt = 0; kt < 64; ++kt) {
        const int cur = kt & 1;

        // Issue global loads for the NEXT B tile before computing this one.
        uint4 nH0, nH1, nL0, nL1;
        const bool havenext = (kt < 63);
        if (havenext) {
            const uint4* gH = reinterpret_cast<const uint4*>(bpH + (kt + 1) * 4096);
            const uint4* gL = reinterpret_cast<const uint4*>(bpL + (kt + 1) * 4096);
            nH0 = gH[tid];
            nH1 = gH[tid + 256];
            nL0 = gL[tid];
            nL1 = gL[tid + 256];
        }

        // Three independent WMMA accumulator chains.
        v8f acc0 = {}, acc1 = {}, acc2 = {};
        const unsigned short* cbH = bH[cur];
        const unsigned short* cbL = bL[cur];
#pragma unroll
        for (int cs = 0; cs < 8; ++cs) {
            v16bf bhi = *reinterpret_cast<const v16bf*>(cbH + (cs * 32 + lane) * 16);
            v16bf blo = *reinterpret_cast<const v16bf*>(cbL + (cs * 32 + lane) * 16);
            acc0 = __builtin_amdgcn_wmma_f32_16x16x32_bf16(
                false, ahi[cs].v, false, bhi, (short)0, acc0, false, false);
            acc1 = __builtin_amdgcn_wmma_f32_16x16x32_bf16(
                false, alo[cs].v, false, bhi, (short)0, acc1, false, false);
            acc2 = __builtin_amdgcn_wmma_f32_16x16x32_bf16(
                false, ahi[cs].v, false, blo, (short)0, acc2, false, false);
        }

        // Store next tile into the other LDS buffer while WMMAs drain.
        if (havenext) {
            uint4* sH = reinterpret_cast<uint4*>(bH[cur ^ 1]);
            uint4* sL = reinterpret_cast<uint4*>(bL[cur ^ 1]);
            sH[tid]       = nH0;
            sH[tid + 256] = nH1;
            sL[tid]       = nL0;
            sL[tid + 256] = nL1;
        }

        // D layout: lane&15 = code column n; token row M = (lane>>4)*8 + v.
        const int code = kt * 16 + (lane & 15);
        const float wq = swsq[code];
#pragma unroll
        for (int v = 0; v < 8; ++v) {
            float dot = (acc0[v] + acc1[v]) + acc2[v];
            float s = wq - 2.0f * dot;
            if (s < bestS[v]) { bestS[v] = s; bestI[v] = code; }
        }

        __syncthreads();   // next-tile stores visible; current tile consumed
    }

    // Argmin across the 16 lanes of each N-group (xor 8,4,2,1 stays in-group).
#pragma unroll
    for (int off = 8; off >= 1; off >>= 1) {
#pragma unroll
        for (int v = 0; v < 8; ++v) {
            float so = __shfl_xor(bestS[v], off, 32);
            int   io = __shfl_xor(bestI[v], off, 32);
            if (so < bestS[v] || (so == bestS[v] && io < bestI[v])) {
                bestS[v] = so; bestI[v] = io;
            }
        }
    }

    if ((lane & 15) == 0) {
        const int g = lane >> 4;
#pragma unroll
        for (int v = 0; v < 8; ++v) {
            int token = t0 + wave * 16 + g * 8 + v;
            idxOut[token] = bestI[v];
        }
    }
}

// ---------------------------------------------------------------------------
// Epilogue: gather winning codebook rows, write z_q (NCHW, same strides as z),
// write indices as float, accumulate (1+beta)*MSE into the loss scalar.
// Block = 256 threads handles 32 tokens x 256 channels (coalesced in token).
// ---------------------------------------------------------------------------
__global__ void __launch_bounds__(256)
vq_writeout(const float* __restrict__ z,
            const float* __restrict__ codebook,
            const int* __restrict__ idxWs,
            float* __restrict__ outZq,
            float* __restrict__ outIdx,
            float* __restrict__ outLoss) {
    __shared__ int   sidx[32];
    __shared__ float sred[8];
    const int tid = threadIdx.x;
    const int t0  = blockIdx.x * 32;
    const int b   = t0 >> 10;
    const int s0  = t0 & 1023;

    if (tid < 32) {
        int si = idxWs[t0 + tid];
        sidx[tid] = si;
        outIdx[t0 + tid] = (float)si;
    }
    __syncthreads();

    const int m  = tid & 31;
    const int cg = tid >> 5;
    const float* wrow = codebook + (size_t)sidx[m] * CC;
    float local = 0.f;
#pragma unroll 4
    for (int i = 0; i < 32; ++i) {
        int c = cg * 32 + i;
        size_t off = ((size_t)(b * CC + c) << 10) + s0 + m;
        float a  = z[off];
        float wv = wrow[c];
        outZq[off] = wv;                 // straight-through output == z_q
        float d = a - wv;
        local += d * d;
    }
#pragma unroll
    for (int o = 16; o >= 1; o >>= 1) local += __shfl_xor(local, o, 32);
    if ((tid & 31) == 0) sred[tid >> 5] = local;
    __syncthreads();
    if (tid == 0) {
        float s = 0.f;
#pragma unroll
        for (int w2 = 0; w2 < 8; ++w2) s += sred[w2];
        atomicAdd(outLoss, s * ((1.0f + BETA) / (float)(NTOK * CC)));
    }
}

// ---------------------------------------------------------------------------
extern "C" void kernel_launch(void* const* d_in, const int* in_sizes, int n_in,
                              void* d_out, int out_size, void* d_ws, size_t ws_size,
                              hipStream_t stream) {
    const float* z        = (const float*)d_in[0];   // [32,256,32,32]
    const float* codebook = (const float*)d_in[1];   // [1024,256]

    float* outZq   = (float*)d_out;                        // 8388608 floats
    float* outIdx  = outZq + (size_t)BB * CC * HH * WW;    // 32768 floats
    float* outLoss = outIdx + NTOK;                        // 1 float

    char* ws = (char*)d_ws;
    unsigned short* bpH  = (unsigned short*)ws;                       // 512 KB
    unsigned short* bpL  = (unsigned short*)(ws + 524288);            // 512 KB
    float*          wsq  = (float*)(ws + 1048576);                    //   4 KB
    int*            idxW = (int*)(ws + 1048576 + 4096);               // 128 KB

    vq_init_loss<<<1, 64, 0, stream>>>(outLoss);
    vq_prep_bpack<<<1024, 256, 0, stream>>>(codebook, bpH, bpL);
    vq_prep_wsq<<<4, 256, 0, stream>>>(codebook, wsq);
    vq_quantize<<<NTOK / 128, 256, 0, stream>>>(z, bpH, bpL, wsq, idxW);
    vq_writeout<<<NTOK / 32, 256, 0, stream>>>(z, codebook, idxW,
                                               outZq, outIdx, outLoss);
}
